// LigerFusedLinearSimPOLoss_84112639525269
// MI455X (gfx1250) — matrix operand order repaired and checked by reading, not compiled, not run on hardware
//
#include <hip/hip_runtime.h>
#include <hip/hip_bf16.h>
#include <math.h>

// ---------------- problem constants (from reference) ----------------
#define VDIM   32000          // vocab
#define HDIM   4096           // hidden
#define TSEQ   512            // tokens per sequence
#define NSEQ   16             // sequences (first 8 chosen, last 8 rejected)
#define NTOK   (NSEQ * TSEQ)  // 8192 tokens
#define IGNORE_INDEX (-100)
#define BETA   0.1f
#define ALPHA  1.0f
#define GAMMA  0.5f

// ---------------- tiling ----------------
#define MT     32                 // token rows per workgroup (2 x 16-row WMMA subtiles)
#define TPB    256                // 8 waves (wave32)
#define NWAVE  8
#define SA     (HDIM + 8)         // padded LDS row stride in bf16 elems (16B skew -> no bank conflicts)

typedef __attribute__((ext_vector_type(16))) __bf16 v16bf;
typedef __attribute__((ext_vector_type(8)))  __bf16 v8bf;
typedef __attribute__((ext_vector_type(4)))  __bf16 v4bf;
typedef __attribute__((ext_vector_type(8)))  float  v8f;

// =====================================================================
// Kernel 0 (optional): one-shot W f32 -> bf16 into workspace.
// Halves the HBM weight stream for every GEMM workgroup afterwards
// (134 GB -> 67 GB across the 256 M-tiles; 262 MB also L2-friendlier),
// and strips all cvt VALU + half the loads from the WMMA inner loop.
// =====================================================================
__global__ __launch_bounds__(256)
void wconv_kernel(const float* __restrict__ W, __bf16* __restrict__ Wbf)
{
    const size_t n8  = (size_t)VDIM * HDIM / 8;
    const size_t gsz = (size_t)gridDim.x * 256;
    for (size_t i = blockIdx.x * 256ull + threadIdx.x; i < n8; i += gsz) {
        const float4* src = (const float4*)(W + i * 8);
        float4 a = src[0], b = src[1];
        union { v8bf v; __bf16 e[8]; } p;
        p.e[0] = (__bf16)a.x; p.e[1] = (__bf16)a.y;
        p.e[2] = (__bf16)a.z; p.e[3] = (__bf16)a.w;
        p.e[4] = (__bf16)b.x; p.e[5] = (__bf16)b.y;
        p.e[6] = (__bf16)b.z; p.e[7] = (__bf16)b.w;
        *(v8bf*)(Wbf + i * 8) = p.v;
    }
}

// =====================================================================
// Label-logit kernel: lbl[tok] = X[tok] . W[label] + bias[label], in f32.
// 8 waves/block, one token per wave; removes all label-capture branches
// from the GEMM epilogue and is MORE accurate (pure f32 dot).
// =====================================================================
__global__ __launch_bounds__(TPB)
void label_logit_kernel(const float* __restrict__ W,
                        const float* __restrict__ X,
                        const int*   __restrict__ tgt,
                        const float* __restrict__ bias,
                        float* __restrict__ lbl)
{
    const int wave = threadIdx.x >> 5;
    const int lane = threadIdx.x & 31;
    const int tok  = blockIdx.x * NWAVE + wave;
    const int lab  = tgt[tok];
    if (lab == IGNORE_INDEX) {              // masked: value never used (mask=0)
        if (lane == 0) lbl[tok] = 0.0f;
        return;
    }
    const float4* xr = (const float4*)(X + (size_t)tok * HDIM);
    const float4* wr = (const float4*)(W + (size_t)lab * HDIM);
    float s = 0.0f;
    for (int i = lane; i < HDIM / 4; i += 32) {
        float4 a = xr[i], b = wr[i];
        s = fmaf(a.x, b.x, s);
        s = fmaf(a.y, b.y, s);
        s = fmaf(a.z, b.z, s);
        s = fmaf(a.w, b.w, s);
    }
#pragma unroll
    for (int off = 16; off >= 1; off >>= 1) s += __shfl_xor(s, off);
    if (lane == 0) lbl[tok] = s + bias[lab];
}

// =====================================================================
// Kernel 1: fused  logits = X*W^T + bias  ->  online row max / sumexp.
// Never stores logits.  PRE=true reads pre-converted bf16 weights.
// =====================================================================
template <bool PRE>
__global__ __launch_bounds__(TPB)
void simpo_gemm_lse_kernel(const float*  __restrict__ W,
                           const __bf16* __restrict__ Wbf,
                           const float*  __restrict__ X,
                           const float*  __restrict__ bias,
                           float* __restrict__ lse_out)   // [NTOK] logsumexp
{
    extern __shared__ char smem[];
    __bf16* Ash  = (__bf16*)smem;                                   // MT x SA bf16
    float*  runm = (float*)(smem + (size_t)MT * SA * 2);            // NWAVE x MT
    float*  runs = runm + NWAVE * MT;                               // NWAVE x MT

    const int tid  = threadIdx.x;
    const int wave = tid >> 5;
    const int lane = tid & 31;
    const int lcol = lane & 15;       // column within 16-wide subtile / A row
    const int half = lane >> 4;       // which 16-lane half
    const int tok0 = blockIdx.x * MT;

    // ---- stage A tile: f32 -> bf16 into LDS (row-major, padded stride) ----
    for (int r = 0; r < MT; ++r) {
        const float4* src = (const float4*)(X + (size_t)(tok0 + r) * HDIM);
        __bf16* dst = Ash + (size_t)r * SA;
        for (int k4 = tid; k4 < HDIM / 4; k4 += TPB) {
            float4 f = src[k4];
            v4bf p;
            p[0] = (__bf16)f.x; p[1] = (__bf16)f.y;
            p[2] = (__bf16)f.z; p[3] = (__bf16)f.w;
            *(v4bf*)(dst + 4 * k4) = p;
        }
    }
    __syncthreads();

    // per-lane running softmax stats, one per (subtile, vgpr) slot
    float rm[16], rs[16];
#pragma unroll
    for (int i = 0; i < 16; ++i) { rm[i] = -__builtin_inff(); rs[i] = 0.0f; }

    const int ako = half * 16;   // A frag K offset (contiguous K, matches B)
    const int bko = half * 16;   // B frag K offset (contiguous K, matches A)

    for (int vb = wave * 16; vb < VDIM; vb += NWAVE * 16) {
        v8f acc[2];
        acc[0] = (v8f){0.f, 0.f, 0.f, 0.f, 0.f, 0.f, 0.f, 0.f};
        acc[1] = acc[0];

        const float*  wrow  = W   + (size_t)(vb + lcol) * HDIM + bko;
        const __bf16* wrowb = Wbf + (size_t)(vb + lcol) * HDIM + bko;
        const float   bv    = bias[vb + lcol];

        for (int kc = 0; kc < HDIM; kc += 32) {
            // ---- B fragment (16 K-contiguous values of one weight row) ----
            union { v16bf v; v8bf h[2]; __bf16 e[16]; } B;
            if constexpr (PRE) {
                __builtin_prefetch(wrowb + kc + 512, 0, 1);
                B.h[0] = *(const v8bf*)(wrowb + kc);
                B.h[1] = *(const v8bf*)(wrowb + kc + 8);
            } else {
                __builtin_prefetch(wrow + kc + 256, 0, 1);
                const float4* bp = (const float4*)(wrow + kc);
                float4 q0 = bp[0], q1 = bp[1], q2 = bp[2], q3 = bp[3];
                float bf[16] = {q0.x, q0.y, q0.z, q0.w, q1.x, q1.y, q1.z, q1.w,
                                q2.x, q2.y, q2.z, q2.w, q3.x, q3.y, q3.z, q3.w};
#pragma unroll
                for (int i = 0; i < 16; ++i) B.e[i] = (__bf16)bf[i];
            }

            // ---- A fragments for the two 16-row subtiles (from LDS) ----
            union { v16bf v; v8bf h[2]; } A0, A1;
            const __bf16* a0p = Ash + (size_t)lcol * SA + kc + ako;
            A0.h[0] = *(const v8bf*)(a0p);
            A0.h[1] = *(const v8bf*)(a0p + 8);
            const __bf16* a1p = Ash + (size_t)(lcol + 16) * SA + kc + ako;
            A1.h[0] = *(const v8bf*)(a1p);
            A1.h[1] = *(const v8bf*)(a1p + 8);

            acc[0] = __builtin_amdgcn_wmma_f32_16x16x32_bf16(
                false, A0.v, false, B.v, (short)0, acc[0], false, false);
            acc[1] = __builtin_amdgcn_wmma_f32_16x16x32_bf16(
                false, A1.v, false, B.v, (short)0, acc[1], false, false);
        }

        // ---- epilogue: bias + lane-local online softmax update only ----
#pragma unroll
        for (int s = 0; s < 2; ++s) {
#pragma unroll
            for (int r = 0; r < 8; ++r) {
                const int slot = s * 8 + r;
                float x  = acc[s][r] + bv;
                float m  = rm[slot];
                float nm = fmaxf(m, x);
                rs[slot] = rs[slot] * __expf(m - nm) + __expf(x - nm);
                rm[slot] = nm;
            }
        }
    }

    // ---- one-time cross-lane merge (16 lanes of each half; xor<16) ----
#pragma unroll
    for (int s = 0; s < 2; ++s) {
#pragma unroll
        for (int r = 0; r < 8; ++r) {
            const int slot = s * 8 + r;
            float m = rm[slot], sv = rs[slot];
#pragma unroll
            for (int off = 1; off < 16; off <<= 1) {
                float om = __shfl_xor(m, off);
                float os = __shfl_xor(sv, off);
                float nm = fmaxf(m, om);
                sv = sv * __expf(m - nm) + os * __expf(om - nm);
                m  = nm;
            }
            if (lcol == 0) {  // lanes 0 & 16 active; distinct rows -> no race
                int rl = s * 16 + r + 8 * half;
                runm[wave * MT + rl] = m;
                runs[wave * MT + rl] = sv;
            }
        }
    }

    __syncthreads();
    // ---- merge the 8 per-wave stats -> per-token logsumexp ----
    if (tid < MT) {
        float m = -__builtin_inff(), sv = 0.0f;
        for (int w = 0; w < NWAVE; ++w) {
            float wm = runm[w * MT + tid];
            float ws = runs[w * MT + tid];
            float nm = fmaxf(m, wm);
            sv = sv * __expf(m - nm) + ws * __expf(wm - nm);
            m  = nm;
        }
        lse_out[tok0 + tid] = m + __logf(sv);
    }
}

// =====================================================================
// Kernel 2: per-sequence averages -> SimPO loss scalar.
// =====================================================================
__global__ __launch_bounds__(256)
void simpo_reduce_kernel(const int*   __restrict__ tgt,
                         const float* __restrict__ lbl,
                         const float* __restrict__ lse,
                         float* __restrict__ out)
{
    __shared__ float ssum[NSEQ];
    __shared__ int   scnt[NSEQ];
    const int tid = threadIdx.x;
    if (tid < NSEQ) { ssum[tid] = 0.0f; scnt[tid] = 0; }
    __syncthreads();

    for (int t = tid; t < NTOK; t += 256) {
        int  lab = tgt[t];
        bool msk = (lab != IGNORE_INDEX);
        float lp = msk ? (lbl[t] - lse[t]) : 0.0f;
        int  s   = t >> 9;                 // t / TSEQ
        atomicAdd(&ssum[s], lp);
        if (msk) atomicAdd(&scnt[s], 1);
    }
    __syncthreads();

    if (tid == 0) {
        float avg[NSEQ];
        float csum = 0.0f; int ccnt = 0;
        for (int s = 0; s < NSEQ; ++s) {
            int c = scnt[s] > 1 ? scnt[s] : 1;
            avg[s] = ssum[s] / (float)c;
            if (s < NSEQ / 2) { csum += ssum[s]; ccnt += scnt[s]; }
        }
        if (ccnt < 1) ccnt = 1;
        float nll = -csum / (float)ccnt;
        float asum = 0.0f;
        for (int i = 0; i < NSEQ / 2; ++i) {
            float p = BETA * (avg[i] - avg[i + NSEQ / 2]) - GAMMA;
            // numerically stable log_sigmoid
            float lsig = (p < 0.0f) ? (p - log1pf(__expf(p)))
                                    : (-log1pf(__expf(-p)));
            asum += lsig;
        }
        out[0] = ALPHA * nll - asum / (float)(NSEQ / 2);
    }
}

// =====================================================================
extern "C" void kernel_launch(void* const* d_in, const int* in_sizes, int n_in,
                              void* d_out, int out_size, void* d_ws, size_t ws_size,
                              hipStream_t stream) {
    (void)in_sizes; (void)n_in; (void)out_size;
    const float* W    = (const float*)d_in[0];   // lin_weight [V,H]
    const float* X    = (const float*)d_in[1];   // _input     [B2,T,H]
    const int*   tgt  = (const int*)d_in[2];     // target     [B2,T]
    const float* bias = (const float*)d_in[3];   // bias       [V]

    float*  lbl = (float*)d_ws;                  // [NTOK] label logits
    float*  lse = lbl + NTOK;                    // [NTOK] logsumexp
    size_t  statBytes = 2ull * NTOK * sizeof(float);
    __bf16* wbf = (__bf16*)((char*)d_ws + statBytes);
    const bool pre = ws_size >= statBytes + (size_t)VDIM * HDIM * 2;

    size_t shmem = (size_t)MT * SA * 2 /*bf16 A tile*/
                 + (size_t)NWAVE * MT * 2 * sizeof(float) /*stats*/;

    label_logit_kernel<<<NTOK / NWAVE, TPB, 0, stream>>>(W, X, tgt, bias, lbl);
    if (pre) {
        wconv_kernel<<<2048, 256, 0, stream>>>(W, wbf);
        simpo_gemm_lse_kernel<true><<<NTOK / MT, TPB, shmem, stream>>>(
            W, wbf, X, bias, lse);
    } else {
        simpo_gemm_lse_kernel<false><<<NTOK / MT, TPB, shmem, stream>>>(
            W, (const __bf16*)nullptr, X, bias, lse);
    }
    simpo_reduce_kernel<<<1, 256, 0, stream>>>(tgt, lbl, lse, (float*)d_out);
}
